// SC_54485955117063
// MI455X (gfx1250) — compile-verified, matrix operand order checked by reading, not run on hardware
//
#include <hip/hip_runtime.h>
#include <hip/hip_bf16.h>
#include <stdint.h>

// ---------------- CDNA5 WMMA types / helpers ----------------
typedef __attribute__((ext_vector_type(16))) __bf16 v16bf;
typedef __attribute__((ext_vector_type(8)))  float  v8f;
typedef unsigned short u16;
typedef unsigned int   u32;

union FragU { uint4 q[2]; v16bf v; };

// Load a 16-element bf16 fragment: two contiguous 16B runs, p and p+16 elems.
// Matches the 16-bit A/B operand layout: lane L holds row L%16 and
// K in {g*8..g*8+7} U {16+g*8..16+g*8+7}, g = L/16.  (ISA 7.12.2)
__device__ __forceinline__ v16bf ldfrag(const u16* __restrict__ p) {
  FragU f;
  f.q[0] = *reinterpret_cast<const uint4*>(p);
  f.q[1] = *reinterpret_cast<const uint4*>(p + 16);
  return f.v;
}

__device__ __forceinline__ v8f wmma_bf16(v16bf a, v16bf b, v8f c) {
  // D = A(16x32) * B(32x16) + C, f32 accumulate
  return __builtin_amdgcn_wmma_f32_16x16x32_bf16(
      false, a, false, b, (short)0, c, false, false);
}

__device__ __forceinline__ u16 f2bf(float f) {
  union { float f; u32 u; } v; v.f = f;
  u32 u = v.u;
  return (u16)((u + 0x7FFFu + ((u >> 16) & 1u)) >> 16);  // RNE
}

// ---------------- problem constants ----------------
#define BETA 0.95122942450071400910f   // exp(-1/20)

constexpr int T    = 100;
constexpr int NB   = 32;     // batch
constexpr int DIN  = 2;      // input channels
constexpr int HIN  = 32;
constexpr int C1   = 32, H1 = 28;
constexpr int C2   = 64, H2 = 24;
constexpr int NREC = 512, NCLS = 11;
constexpr int KIN  = C2 * H2 * H2;          // 36864
constexpr int P1   = H1 * H1;               // 784 pixels
constexpr int P2   = H2 * H2;               // 576 pixels
constexpr int N1   = NB * P1 * C1;          // m1/s1 elems (channel-last)
constexpr int N2   = NB * KIN;              // m2/s2 elems ([b][c][y][x] flat)
constexpr int SL   = 36;                    // fc1 split-K slices
constexpr int CPS  = (KIN / 32) / SL;       // 32 K-chunks per slice

// ---------------- workspace layout (f32-elem offsets for states) ----------------
constexpr size_t OF_M1   = 0;
constexpr size_t OF_M2   = OF_M1 + N1;
constexpr size_t OF_MR   = OF_M2 + N2;
constexpr size_t OF_MR2  = OF_MR + NB * NREC;
constexpr size_t OF_MO   = OF_MR2 + NB * NREC;          // padded [32][16]
constexpr size_t STATE_F32 = OF_MO + NB * 16;

constexpr size_t alignup(size_t x) { return (x + 255) & ~size_t(255); }
constexpr size_t BY_PART = alignup(STATE_F32 * 4);                       // SL*32*512 f32
constexpr size_t BY_S1   = alignup(BY_PART + (size_t)SL * NB * NREC * 4);// s1 bf16 [b][y][x][c]
constexpr size_t BY_S2   = alignup(BY_S1 + (size_t)N1 * 2);              // s2 bf16 [b][c*576+p]
constexpr size_t BY_SR   = alignup(BY_S2 + (size_t)N2 * 2);
constexpr size_t BY_SR2  = alignup(BY_SR + (size_t)NB * NREC * 2);
constexpr size_t BY_WC2  = alignup(BY_SR2 + (size_t)NB * NREC * 2);      // [64][800] bf16
constexpr size_t BY_WIN  = alignup(BY_WC2 + (size_t)C2 * 800 * 2);       // [512][36864] bf16
constexpr size_t BY_WIN2 = alignup(BY_WIN + (size_t)NREC * KIN * 2);     // [512][512] bf16
constexpr size_t BY_WOUT = alignup(BY_WIN2 + (size_t)NREC * NREC * 2);   // [16][512] bf16

// ---------------- prep kernels (once per launch) ----------------
__global__ __launch_bounds__(256) void k_zero(float* __restrict__ p, int n) {
  int i = blockIdx.x * 256 + threadIdx.x;
  if (i < n) p[i] = 0.0f;
}

__global__ __launch_bounds__(256) void k_cvt(u16* __restrict__ dst,
                                             const float* __restrict__ src, int n) {
  int i = blockIdx.x * 256 + threadIdx.x;
  if (i < n) dst[i] = f2bf(src[i]);
}

// wb_c2[n][(fy*5+fx)*32 + c] = w_conv2[n][c][fy][fx]
__global__ __launch_bounds__(256) void k_prep_wc2(u16* __restrict__ dst,
                                                  const float* __restrict__ w) {
  int i = blockIdx.x * 256 + threadIdx.x;
  if (i >= C2 * 800) return;
  int n = i / 800, k = i % 800;
  int kc = k >> 5, c = k & 31, fy = kc / 5, fx = kc % 5;
  dst[i] = f2bf(w[((n * 32 + c) * 5 + fy) * 5 + fx]);
}

// wb_out[n][k], rows 11..15 zero-padded
__global__ __launch_bounds__(256) void k_prep_wout(u16* __restrict__ dst,
                                                   const float* __restrict__ w) {
  int i = blockIdx.x * 256 + threadIdx.x;
  if (i >= 16 * NREC) return;
  int n = i / NREC, k = i % NREC;
  dst[i] = (n < NCLS) ? f2bf(w[n * NREC + k]) : (u16)0;
}

// ---------------- per-timestep kernels ----------------
// conv1 (direct, VALU) + leaky; s1 written channel-last bf16 [b][y][x][c]
__global__ __launch_bounds__(256) void k_conv1(const float* __restrict__ x,
                                               const float* __restrict__ w,
                                               const float* __restrict__ bias,
                                               float* __restrict__ m1,
                                               u16* __restrict__ s1) {
  int idx = blockIdx.x * 256 + threadIdx.x;           // N1 = 802816
  int c = idx & 31;
  int t = idx >> 5;
  int xx = t % H1; t /= H1;
  int yy = t % H1; t /= H1;
  int b = t;
  const float* xb = x + (size_t)b * DIN * HIN * HIN + yy * HIN + xx;
  const float* wc = w + c * (DIN * 25);
  float acc = bias[c];
#pragma unroll
  for (int d = 0; d < DIN; ++d) {
#pragma unroll
    for (int fy = 0; fy < 5; ++fy)
#pragma unroll
      for (int fx = 0; fx < 5; ++fx)
        acc = fmaf(xb[d * HIN * HIN + fy * HIN + fx], wc[d * 25 + fy * 5 + fx], acc);
  }
  float mem = m1[idx];
  float reset = (mem > 1.0f) ? 1.0f : 0.0f;
  mem = BETA * mem + acc - reset;
  s1[idx] = (mem > 1.0f) ? (u16)0x3F80 : (u16)0;      // bf16 1.0 / 0.0
  m1[idx] = mem;
}

// conv2 implicit GEMM via WMMA bf16 + leaky, register-blocked 2 M-tiles/wave.
// grid (18 mtile-pairs, 32 batch), 128 thr = 4 waves, wave -> out-channel tile.
// All filter-tap offsets are compile-time immediates off per-lane base pointers.
__global__ __launch_bounds__(128) void k_conv2(const u16* __restrict__ s1,
                                               const u16* __restrict__ wb,
                                               const float* __restrict__ bias,
                                               float* __restrict__ m2,
                                               u16* __restrict__ s2) {
  const int mt2 = blockIdx.x, b = blockIdx.y;
  const int wv = threadIdx.x >> 5, lane = threadIdx.x & 31;
  const int g = lane >> 4, l = lane & 15;
  const int n = wv * 16 + l;                 // out channel (B row)
  const int p0 = mt2 * 32;                   // pixel base (two 16-row tiles)
  const int py0 = (p0 + l) / H2,      px0 = (p0 + l) % H2;       // A rows, tile 0
  const int py1 = (p0 + 16 + l) / H2, px1 = (p0 + 16 + l) % H2;  // A rows, tile 1
  const u16* s1b = s1 + (size_t)b * P1 * C1 + g * 8;
  const u16* a0b = s1b + (py0 * H1 + px0) * C1;   // lane base, tile 0
  const u16* a1b = s1b + (py1 * H1 + px1) * C1;   // lane base, tile 1
  const u16* wrow = wb + (size_t)n * 800 + g * 8;
  v8f c0 = {}, c1 = {};
#pragma unroll
  for (int fy = 0; fy < 5; ++fy)
#pragma unroll
    for (int fx = 0; fx < 5; ++fx) {
      const int ao = (fy * H1 + fx) * C1;         // compile-time immediate
      v16bf bm = ldfrag(wrow + (fy * 5 + fx) * 32);
      c0 = wmma_bf16(ldfrag(a0b + ao), bm, c0);
      c1 = wmma_bf16(ldfrag(a1b + ao), bm, c1);
    }
  const float bv = bias[n];
#pragma unroll
  for (int r = 0; r < 8; ++r) {
    // D layout: m = r + 8*g, n = lane%16
    int p = p0 + 8 * g + r;                               // tile 0 pixel
    size_t idx = (size_t)b * KIN + (size_t)n * P2 + p;    // [b][c][y][x] flat
    float mem = m2[idx];
    float reset = (mem > 1.0f) ? 1.0f : 0.0f;
    mem = BETA * mem + (c0[r] + bv) - reset;
    s2[idx] = (mem > 1.0f) ? (u16)0x3F80 : (u16)0;
    m2[idx] = mem;

    size_t idx1 = idx + 16;                               // tile 1 pixel (+16)
    float mem1 = m2[idx1];
    float reset1 = (mem1 > 1.0f) ? 1.0f : 0.0f;
    mem1 = BETA * mem1 + (c1[r] + bv) - reset1;
    s2[idx1] = (mem1 > 1.0f) ? (u16)0x3F80 : (u16)0;
    m2[idx1] = mem1;
  }
}

// fc1: C[32x512] = s2[32x36864] x w_in^T, split-K (no atomics).
// Register-blocked: each wave owns a 32x32 output tile (2 mtiles x 2 ntiles).
// Inner loop: 4 chunks with immediate offsets, then bump 4 base pointers.
__global__ __launch_bounds__(32) void k_fc1(const u16* __restrict__ s2,
                                            const u16* __restrict__ wb,
                                            float* __restrict__ part) {
  const int ng = blockIdx.x, slice = blockIdx.y;   // ng: 0..15, slice: 0..SL-1
  const int lane = threadIdx.x & 31, g = lane >> 4, l = lane & 15;
  const int n0 = ng * 32 + l, n1 = n0 + 16;
  const size_t k0 = (size_t)slice * CPS * 32 + g * 8;
  const u16* pa0 = s2 + (size_t)l * KIN + k0;         // batch rows 0..15
  const u16* pa1 = s2 + (size_t)(16 + l) * KIN + k0;  // batch rows 16..31
  const u16* pb0 = wb + (size_t)n0 * KIN + k0;
  const u16* pb1 = wb + (size_t)n1 * KIN + k0;
  v8f c00 = {}, c01 = {}, c10 = {}, c11 = {};
#pragma unroll 1
  for (int cc = 0; cc < CPS; cc += 4) {
#pragma unroll
    for (int j = 0; j < 4; ++j) {
      const int o = j * 32;                         // compile-time immediate
      v16bf a0 = ldfrag(pa0 + o);
      v16bf a1 = ldfrag(pa1 + o);
      v16bf b0 = ldfrag(pb0 + o);
      v16bf b1 = ldfrag(pb1 + o);
      c00 = wmma_bf16(a0, b0, c00);
      c01 = wmma_bf16(a0, b1, c01);
      c10 = wmma_bf16(a1, b0, c10);
      c11 = wmma_bf16(a1, b1, c11);
    }
    pa0 += 128; pa1 += 128; pb0 += 128; pb1 += 128;
  }
  float* pp = part + (size_t)slice * NB * NREC;
#pragma unroll
  for (int r = 0; r < 8; ++r) {
    int m0 = 8 * g + r, m1r = 16 + 8 * g + r;
    pp[m0 * NREC + n0]  = c00[r];
    pp[m0 * NREC + n1]  = c01[r];
    pp[m1r * NREC + n0] = c10[r];
    pp[m1r * NREC + n1] = c11[r];
  }
}

// reduce fc1 partials + bias + leaky -> sr (bf16), mr
__global__ __launch_bounds__(256) void k_leaky_r(const float* __restrict__ part,
                                                 const float* __restrict__ bias,
                                                 float* __restrict__ mr,
                                                 u16* __restrict__ sr) {
  int i = blockIdx.x * 256 + threadIdx.x;        // 16384
  float acc = bias[i & (NREC - 1)];
#pragma unroll
  for (int s = 0; s < SL; ++s) acc += part[(size_t)s * NB * NREC + i];
  float mem = mr[i];
  float reset = (mem > 1.0f) ? 1.0f : 0.0f;
  mem = BETA * mem + acc - reset;
  sr[i] = (mem > 1.0f) ? (u16)0x3F80 : (u16)0;
  mr[i] = mem;
}

// fc2: C[32x512] = sr x w_in2^T + b, fused leaky -> sr2, mr2.
// Each wave: one ntile, both mtiles (B-frag loaded once, 2 WMMAs/chunk).
__global__ __launch_bounds__(32) void k_fc2(const u16* __restrict__ sr,
                                            const u16* __restrict__ wb,
                                            const float* __restrict__ bias,
                                            float* __restrict__ mr2,
                                            u16* __restrict__ sr2) {
  const int ntile = blockIdx.x;
  const int lane = threadIdx.x & 31, g = lane >> 4, l = lane & 15;
  const int n = ntile * 16 + l;
  const u16* arow0 = sr + (size_t)l * NREC + g * 8;
  const u16* arow1 = sr + (size_t)(16 + l) * NREC + g * 8;
  const u16* brow  = wb + (size_t)n * NREC + g * 8;
  v8f c0 = {}, c1 = {};
#pragma unroll
  for (int kb = 0; kb < NREC; kb += 32) {           // kb: compile-time immediates
    v16bf bm = ldfrag(brow + kb);
    c0 = wmma_bf16(ldfrag(arow0 + kb), bm, c0);
    c1 = wmma_bf16(ldfrag(arow1 + kb), bm, c1);
  }
  const float bv = bias[n];
#pragma unroll
  for (int r = 0; r < 8; ++r) {
#pragma unroll
    for (int h = 0; h < 2; ++h) {
      int idx = (h * 16 + 8 * g + r) * NREC + n;
      float cv = (h == 0) ? c0[r] : c1[r];
      float mem = mr2[idx];
      float reset = (mem > 1.0f) ? 1.0f : 0.0f;
      mem = BETA * mem + (cv + bv) - reset;
      sr2[idx] = (mem > 1.0f) ? (u16)0x3F80 : (u16)0;
      mr2[idx] = mem;
    }
  }
}

// output layer: C[32x16(11)] = sr2 x w_out^T + b, leaky -> write spk/mem traces
__global__ __launch_bounds__(64) void k_out(const u16* __restrict__ sr2,
                                            const u16* __restrict__ wb,
                                            const float* __restrict__ bias,
                                            float* __restrict__ mo,
                                            float* __restrict__ out_spk,
                                            float* __restrict__ out_mem) {
  const int mtile = threadIdx.x >> 5;
  const int lane = threadIdx.x & 31, g = lane >> 4, l = lane & 15;
  const u16* arow = sr2 + (size_t)(mtile * 16 + l) * NREC + g * 8;
  const u16* brow = wb + (size_t)l * NREC + g * 8;     // padded rows give 0
  v8f c = {};
#pragma unroll
  for (int kb = 0; kb < NREC; kb += 32)
    c = wmma_bf16(ldfrag(arow + kb), ldfrag(brow + kb), c);
  const int n = l;
  if (n < NCLS) {
    const float bv = bias[n];
#pragma unroll
    for (int r = 0; r < 8; ++r) {
      int m = mtile * 16 + 8 * g + r;
      int idx = m * 16 + n;
      float mem = mo[idx];
      float reset = (mem > 1.0f) ? 1.0f : 0.0f;
      mem = BETA * mem + (c[r] + bv) - reset;
      float sp = (mem > 1.0f) ? 1.0f : 0.0f;
      out_spk[m * NCLS + n] = sp;
      out_mem[m * NCLS + n] = mem;
      mo[idx] = mem;
    }
  }
}

// ---------------- host-side launch ----------------
extern "C" void kernel_launch(void* const* d_in, const int* in_sizes, int n_in,
                              void* d_out, int out_size, void* d_ws, size_t ws_size,
                              hipStream_t stream) {
  const float* x        = (const float*)d_in[0];
  const float* w_conv1  = (const float*)d_in[1];
  const float* b_conv1  = (const float*)d_in[2];
  const float* w_conv2  = (const float*)d_in[3];
  const float* b_conv2  = (const float*)d_in[4];
  const float* w_in     = (const float*)d_in[5];
  const float* b_in     = (const float*)d_in[6];
  const float* w_in2    = (const float*)d_in[7];
  const float* b_in2    = (const float*)d_in[8];
  const float* w_out    = (const float*)d_in[9];
  const float* b_out    = (const float*)d_in[10];

  char*  ws  = (char*)d_ws;
  float* wsf = (float*)d_ws;
  float* m1  = wsf + OF_M1;
  float* m2  = wsf + OF_M2;
  float* mr  = wsf + OF_MR;
  float* mr2 = wsf + OF_MR2;
  float* mo  = wsf + OF_MO;
  float* part = (float*)(ws + BY_PART);
  u16* s1   = (u16*)(ws + BY_S1);
  u16* s2   = (u16*)(ws + BY_S2);
  u16* sr   = (u16*)(ws + BY_SR);
  u16* sr2  = (u16*)(ws + BY_SR2);
  u16* wc2  = (u16*)(ws + BY_WC2);
  u16* win  = (u16*)(ws + BY_WIN);
  u16* win2 = (u16*)(ws + BY_WIN2);
  u16* wout = (u16*)(ws + BY_WOUT);

  float* out_spk = (float*)d_out;                       // [T][32][11]
  float* out_mem = (float*)d_out + (size_t)T * NB * NCLS;

  // ---- once per launch: zero states, convert weights to bf16 ----
  k_zero<<<((int)STATE_F32 + 255) / 256, 256, 0, stream>>>(wsf, (int)STATE_F32);
  k_cvt<<<(NREC * KIN + 255) / 256, 256, 0, stream>>>(win, w_in, NREC * KIN);
  k_cvt<<<(NREC * NREC + 255) / 256, 256, 0, stream>>>(win2, w_in2, NREC * NREC);
  k_prep_wc2<<<(C2 * 800 + 255) / 256, 256, 0, stream>>>(wc2, w_conv2);
  k_prep_wout<<<(16 * NREC + 255) / 256, 256, 0, stream>>>(wout, w_out);

  // ---- scan over timesteps ----
  for (int t = 0; t < T; ++t) {
    const float* xt = x + (size_t)t * NB * DIN * HIN * HIN;
    k_conv1<<<N1 / 256, 256, 0, stream>>>(xt, w_conv1, b_conv1, m1, s1);
    k_conv2<<<dim3(P2 / 32, NB), 128, 0, stream>>>(s1, wc2, b_conv2, m2, s2);
    k_fc1<<<dim3(NREC / 32, SL), 32, 0, stream>>>(s2, win, part);
    k_leaky_r<<<(NB * NREC) / 256, 256, 0, stream>>>(part, b_in, mr, sr);
    k_fc2<<<NREC / 16, 32, 0, stream>>>(sr, win2, b_in2, mr2, sr2);
    k_out<<<1, 64, 0, stream>>>(sr2, wout, b_out, mo,
                                out_spk + (size_t)t * NB * NCLS,
                                out_mem + (size_t)t * NB * NCLS);
  }
}
